// TLSTM_Hawkes_3152505995668
// MI455X (gfx1250) — compile-verified, matrix operand order checked by reading, not run on hardware
//
#include <hip/hip_runtime.h>
#include <hip/hip_bf16.h>
#include <stddef.h>

// ---------------------------------------------------------------------------
// TLSTM + GRU + Hawkes-attention head for MI455X (gfx1250, wave32, WMMA).
//
//  * bf16 v_wmma_f32_16x16x32_bf16 for every GEMM (f32 accumulate).
//  * Persistent scan kernels: 16 workgroups x 16-row batch tiles; hh/cc state
//    in LDS; weights stream from L2 every step (hot in 192MB L2). Weight
//    pointers are made opaque per iteration so LICM cannot hoist+spill, and
//    weight fragments are loaded through addrspace(1) pointers (single cast
//    per fragment, pointer arithmetic kept in addrspace(1) so both 16B loads
//    fold into one base + immediate offsets) -> global_load_b128, LOADcnt
//    only, no flat/DScnt coupling, no per-fragment v_add_nc_u64 + v_nop
//    hazard sequences in the WMMA stream.
//  * Input GEMMs (inputs@U_all^T, tl@W_ih^T) fused into the scans to avoid
//    ~0.9GB of f32 intermediate traffic.
//  * global_prefetch of the next step's activation tile overlaps the
//    post-barrier staging load with current-step WMMA work.
// ---------------------------------------------------------------------------

typedef __bf16 bf16_t;
typedef __attribute__((ext_vector_type(16))) __bf16 v16bf;
typedef __attribute__((ext_vector_type(8)))  __bf16 v8bf;
typedef __attribute__((ext_vector_type(8)))  float  v8f;

// global-addrspace pointers
typedef const bf16_t __attribute__((address_space(1)))* gbfp;
typedef const v8bf   __attribute__((address_space(1)))* gv8bfp;

#define B_  256
#define S_  512
#define D_  256
#define H_  256
#define T_  2
#define G4  1024   // 4*H
#define G3  768    // 3*H
#define BT  16     // batch rows per workgroup (== WMMA M)

union BFrag { v16bf v; v8bf h[2]; };

// Defeat LICM: re-materialize the pointer each loop iteration so weight
// loads stay inside the scan loop (L2 streaming) instead of being hoisted
// and spilled to scratch.
__device__ __forceinline__ const bf16_t* opaque(const bf16_t* p) {
  asm volatile("" : "+s"(p));
  return p;
}

// Fragment layout (16x32 bf16 operand, A-style; B mirrored with N for M):
//   lane<16 : row=lane,    K = kbase+{0..7, 16..23}
//   lane>=16: row=lane-16, K = kbase+{8..15,24..31}
// Two 16B loads per lane.

// LDS version (generic pointer; addrspace(3) inferred -> ds_load_b128)
__device__ __forceinline__ v16bf load_frag(const bf16_t* __restrict__ base,
                                           int ld, int kbase, int lane) {
  const int r   = lane & 15;
  const int sel = lane >> 4;
  const bf16_t* p = base + (size_t)r * ld + kbase + sel * 8;
  BFrag f;
  f.h[0] = *(const v8bf*)(p);
  f.h[1] = *(const v8bf*)(p + 16);
  return f.v;
}

// Global version (explicit addrspace(1) -> global_load_b128; pointer math
// stays in addrspace(1) so both loads fold into base + immediate offset)
__device__ __forceinline__ v16bf load_frag_g(const bf16_t* base,
                                             int ld, int kbase, int lane) {
  const int r   = lane & 15;
  const int sel = lane >> 4;
  gbfp p = (gbfp)(unsigned long long)(base + (size_t)r * ld + kbase + sel * 8);
  BFrag f;
  f.h[0] = *(gv8bfp)(p);
  f.h[1] = *(gv8bfp)(p + 16);
  return f.v;
}

__device__ __forceinline__ v8f wmma_bf16(v16bf a, v16bf b, v8f c) {
  return __builtin_amdgcn_wmma_f32_16x16x32_bf16(
      /*neg_a=*/false, a, /*neg_b=*/false, b,
      /*c_mod=*/(short)0, c, /*reuse_a=*/false, /*reuse_b=*/false);
}

__device__ __forceinline__ float sigmoidf_(float x) {
  return 1.0f / (1.0f + __expf(-x));
}

// ------------------------------ TLSTM scan ---------------------------------
// gates = hh@W_all^T + b_Wall + (x@U_all^T + b_Uall)   [fused u computation]
// c_s1  = tanh(cc@W_d^T + b_Wd)
// c_adj = cc - c_s1 + c_s1*ts ;  cc' = sig(f)*c_adj + sig(i)*sig(ct)
// hh'   = sig(o)*tanh(cc') -> tl[s,b,h] (bf16)
__global__ __launch_bounds__(512, 1)
void tlstm_scan_kernel(const float*  __restrict__ inputs,   // (B,S,D)
                       const float*  __restrict__ ts_inv,   // (B,S)
                       const bf16_t* __restrict__ Wall,     // (4H,H) bf16
                       const float*  __restrict__ Wall_b,   // (4H)
                       const bf16_t* __restrict__ Uall,     // (4H,D) bf16
                       const float*  __restrict__ Uall_b,   // (4H)
                       const bf16_t* __restrict__ Wd,       // (H,H) bf16
                       const float*  __restrict__ Wd_b,     // (H)
                       bf16_t*       __restrict__ tl)       // (S,B,H) bf16
{
  __shared__ __align__(16) bf16_t sh_hh[BT * H_];   // h state, bf16
  __shared__ __align__(16) bf16_t sh_cb[BT * H_];   // c state, bf16 (GEMM A)
  __shared__ __align__(16) float  sh_cc[BT * H_];   // c state, f32 (exact)
  __shared__ __align__(16) bf16_t sh_x [BT * D_];   // input row tile
  __shared__ float sh_ts[BT];

  const int tid  = threadIdx.x;
  const int lane = tid & 31;
  const int wave = tid >> 5;            // 0..15, owns h-cols [16w,16w+16)
  const int b0   = blockIdx.x * BT;
  const int hi   = lane >> 4;
  const int r    = lane & 15;
  const int nb   = wave * 16;
  const int col  = nb + r;

  // hoist per-lane biases (combined W_all_b + U_all_b)
  float bias[4];
  for (int gb = 0; gb < 4; ++gb)
    bias[gb] = Wall_b[gb * H_ + col] + Uall_b[gb * H_ + col];
  const float bwd = Wd_b[col];

  for (int i = tid; i < BT * H_; i += 512) {
    sh_hh[i] = (bf16_t)0.f; sh_cb[i] = (bf16_t)0.f; sh_cc[i] = 0.f;
  }
  __syncthreads();

  for (int s = 0; s < S_; ++s) {
    // stage inputs[b0..b0+15, s, :] as bf16 (f32 HBM read, once per element)
    for (int i = tid; i < BT * D_; i += 512) {
      int br = i >> 8, d = i & (D_ - 1);
      sh_x[i] = (bf16_t)inputs[((size_t)(b0 + br) * S_ + s) * D_ + d];
    }
    if (tid < BT) sh_ts[tid] = ts_inv[(size_t)(b0 + tid) * S_ + s];
    // prefetch next step's input tile into cache (global_prefetch path)
    if (s + 1 < S_) {
      int pi = tid * 8;                       // 512 threads x 8 floats = tile
      int br = pi >> 8, d = pi & (D_ - 1);
      __builtin_prefetch(&inputs[((size_t)(b0 + br) * S_ + (s + 1)) * D_ + d], 0, 1);
    }
    __syncthreads();

    // per-iteration opaque weight pointers (keep loads in-loop, L2-resident)
    const bf16_t* WallI = opaque(Wall);
    const bf16_t* UallI = opaque(Uall);
    const bf16_t* WdI   = opaque(Wd);

    // c_s1 pre-activation: cc @ W_d^T  (K=256 -> 8 WMMA)
    v8f cacc = {};
    const bf16_t* Wdp = WdI + (size_t)nb * H_;
    for (int kk = 0; kk < H_; kk += 32)
      cacc = wmma_bf16(load_frag(sh_cb, H_, kk, lane),
                       load_frag_g(Wdp, H_, kk, lane), cacc);

    // gate pre-activations: x@U_all^T + hh@W_all^T (16 WMMA per gate block)
    v8f g[4];
    for (int gb = 0; gb < 4; ++gb) {
      v8f acc = {};
      const bf16_t* Up = UallI + (size_t)(gb * H_ + nb) * D_;
      for (int kk = 0; kk < D_; kk += 32)
        acc = wmma_bf16(load_frag(sh_x, D_, kk, lane),
                        load_frag_g(Up, D_, kk, lane), acc);
      const bf16_t* Wp = WallI + (size_t)(gb * H_ + nb) * H_;
      for (int kk = 0; kk < H_; kk += 32)
        acc = wmma_bf16(load_frag(sh_hh, H_, kk, lane),
                        load_frag_g(Wp,  H_, kk, lane), acc);
      g[gb] = acc;
    }

    // elementwise on C/D fragment: M = e + 8*hi, N = col
    v8f hhn = {}, ccn = {};
    for (int e = 0; e < 8; ++e) {
      int row    = e + 8 * hi;
      float ts   = sh_ts[row];
      float ccv  = sh_cc[row * H_ + col];
      float cs1  = tanhf(cacc[e] + bwd);
      float cadj = ccv - cs1 + cs1 * ts;
      float fg = sigmoidf_(g[0][e] + bias[0]);
      float ig = sigmoidf_(g[1][e] + bias[1]);
      float og = sigmoidf_(g[2][e] + bias[2]);
      float ct = sigmoidf_(g[3][e] + bias[3]);
      float cnew = fg * cadj + ig * ct;
      ccn[e] = cnew;
      hhn[e] = og * tanhf(cnew);
    }
    __syncthreads();   // every wave finished reading hh/cc/x

    for (int e = 0; e < 8; ++e) {
      int row = e + 8 * hi;
      float hv = hhn[e], cv = ccn[e];
      sh_hh[row * H_ + col] = (bf16_t)hv;
      sh_cc[row * H_ + col] = cv;
      sh_cb[row * H_ + col] = (bf16_t)cv;
      tl[((size_t)s * B_ + (b0 + row)) * H_ + col] = (bf16_t)hv;
    }
    __syncthreads();
  }
}

// ------------------------------- GRU scan ----------------------------------
// xg = x@W_ih^T (fused), hg = hh@W_hh^T
// r=sig(xr+hr) z=sig(xz+hz) n=tanh(xn + r*hn) ; hh' = (1-z)*n + z*hh
__global__ __launch_bounds__(512, 1)
void gru_scan_kernel(const bf16_t* __restrict__ tl,     // (S,B,H) bf16
                     const bf16_t* __restrict__ Wih,    // (3H,H) bf16
                     const float*  __restrict__ b_ih,   // (3H)
                     const bf16_t* __restrict__ Whh,    // (3H,H) bf16
                     const float*  __restrict__ b_hh,   // (3H)
                     bf16_t*       __restrict__ ctx,    // (B,S,H) bf16
                     float*        __restrict__ h_n)    // (B,H) f32
{
  __shared__ __align__(16) bf16_t sh_hh[BT * H_];
  __shared__ __align__(16) float  sh_hf[BT * H_];
  __shared__ __align__(16) bf16_t sh_x [BT * H_];

  const int tid  = threadIdx.x;
  const int lane = tid & 31;
  const int wave = tid >> 5;
  const int b0   = blockIdx.x * BT;
  const int hi   = lane >> 4;
  const int r    = lane & 15;
  const int nb   = wave * 16;
  const int col  = nb + r;

  float bih[3], bhh[3];
  for (int gb = 0; gb < 3; ++gb) {
    bih[gb] = b_ih[gb * H_ + col];
    bhh[gb] = b_hh[gb * H_ + col];
  }

  for (int i = tid; i < BT * H_; i += 512) { sh_hh[i] = (bf16_t)0.f; sh_hf[i] = 0.f; }
  __syncthreads();

  for (int s = 0; s < S_; ++s) {
    for (int i = tid; i < BT * H_; i += 512) {
      int br = i >> 8, hc = i & (H_ - 1);
      sh_x[i] = tl[((size_t)s * B_ + (b0 + br)) * H_ + hc];
    }
    if (s + 1 < S_) {
      int pi = tid * 8;
      int br = pi >> 8, hc = pi & (H_ - 1);
      __builtin_prefetch(&tl[((size_t)(s + 1) * B_ + (b0 + br)) * H_ + hc], 0, 1);
    }
    __syncthreads();

    const bf16_t* WihI = opaque(Wih);
    const bf16_t* WhhI = opaque(Whh);

    v8f ra = {}, za = {}, xn = {}, hn = {};
    for (int kk = 0; kk < H_; kk += 32) {
      v16bf ax = load_frag(sh_x,  H_, kk, lane);
      v16bf ah = load_frag(sh_hh, H_, kk, lane);
      ra = wmma_bf16(ax, load_frag_g(WihI + (size_t)(0 * H_ + nb) * H_, H_, kk, lane), ra);
      ra = wmma_bf16(ah, load_frag_g(WhhI + (size_t)(0 * H_ + nb) * H_, H_, kk, lane), ra);
      za = wmma_bf16(ax, load_frag_g(WihI + (size_t)(1 * H_ + nb) * H_, H_, kk, lane), za);
      za = wmma_bf16(ah, load_frag_g(WhhI + (size_t)(1 * H_ + nb) * H_, H_, kk, lane), za);
      xn = wmma_bf16(ax, load_frag_g(WihI + (size_t)(2 * H_ + nb) * H_, H_, kk, lane), xn);
      hn = wmma_bf16(ah, load_frag_g(WhhI + (size_t)(2 * H_ + nb) * H_, H_, kk, lane), hn);
    }

    v8f hhn = {};
    for (int e = 0; e < 8; ++e) {
      int row = e + 8 * hi;
      float hold = sh_hf[row * H_ + col];
      float rr = sigmoidf_(ra[e] + bih[0] + bhh[0]);
      float zz = sigmoidf_(za[e] + bih[1] + bhh[1]);
      float nn = tanhf(xn[e] + bih[2] + rr * (hn[e] + bhh[2]));
      hhn[e] = (1.0f - zz) * nn + zz * hold;
    }
    __syncthreads();

    for (int e = 0; e < 8; ++e) {
      int row = e + 8 * hi;
      float hv = hhn[e];
      sh_hh[row * H_ + col] = (bf16_t)hv;
      sh_hf[row * H_ + col] = hv;
      ctx[((size_t)(b0 + row) * S_ + s) * H_ + col] = (bf16_t)hv;
      if (s == S_ - 1) h_n[(size_t)(b0 + row) * H_ + col] = hv;
    }
    __syncthreads();
  }
}

// ------------------------- Attention + MLP head ----------------------------
// One block per batch row; tiny FLOPs, plain VALU with LDS reductions.
__global__ __launch_bounds__(256)
void attn_head_kernel(const float*  __restrict__ h_n,        // (B,H)
                      const bf16_t* __restrict__ ctx,        // (B,S,H)
                      const float*  __restrict__ attn_in,    // (H,H)
                      const float*  __restrict__ attn_out,   // (H,2H)
                      const float*  __restrict__ timestamps, // (B,S)
                      const float*  __restrict__ Wb,         // (B)
                      const float*  __restrict__ wd,         // (B)
                      const float*  __restrict__ lin1_w,     // (H,H)
                      const float*  __restrict__ lin1_b,     // (H)
                      const float*  __restrict__ lin2_w,     // (T,H)
                      const float*  __restrict__ lin2_b,     // (T)
                      float*        __restrict__ out)        // (B,T)
{
  __shared__ float shn[H_], sq[H_], sw[S_], smix[H_], so1[H_], so2[H_];
  __shared__ float red[256];

  const int b = blockIdx.x, tid = threadIdx.x;
  const bf16_t* cb = ctx + (size_t)b * S_ * H_;

  shn[tid] = h_n[(size_t)b * H_ + tid];
  __syncthreads();

  // q = attn_in @ h_n
  {
    float acc = 0.f;
    const float* wr = attn_in + (size_t)tid * H_;
    for (int k = 0; k < H_; ++k) acc += wr[k] * shn[k];
    sq[tid] = acc;
  }
  __syncthreads();

  // scores[s] = q . context[b,s,:]
  for (int half = 0; half < 2; ++half) {
    int s = tid + half * 256;
    const bf16_t* cr = cb + (size_t)s * H_;
    float sc = 0.f;
    for (int k = 0; k < H_; ++k) sc += sq[k] * (float)cr[k];
    sw[s] = sc;
  }
  __syncthreads();

  // softmax over S=512
  float m = fmaxf(sw[tid], sw[tid + 256]);
  red[tid] = m; __syncthreads();
  for (int off = 128; off; off >>= 1) {
    if (tid < off) red[tid] = fmaxf(red[tid], red[tid + off]);
    __syncthreads();
  }
  float mx = red[0]; __syncthreads();
  float e0 = expf(sw[tid] - mx), e1 = expf(sw[tid + 256] - mx);
  red[tid] = e0 + e1; __syncthreads();
  for (int off = 128; off; off >>= 1) {
    if (tid < off) red[tid] += red[tid + off];
    __syncthreads();
  }
  float inv = 1.0f / red[0]; __syncthreads();

  // Hawkes reweight: w + relu(w * Wb * exp(-wd * t))
  const float wbv = Wb[b], wdv = wd[b];
  {
    float w0 = e0 * inv, w1 = e1 * inv;
    float bt0 = expf(-wdv * timestamps[(size_t)b * S_ + tid]);
    float bt1 = expf(-wdv * timestamps[(size_t)b * S_ + tid + 256]);
    sw[tid]       = w0 + fmaxf(w0 * (wbv * bt0), 0.f);
    sw[tid + 256] = w1 + fmaxf(w1 * (wbv * bt1), 0.f);
  }
  __syncthreads();

  // mix[h] = sum_s w[s] * context[b,s,h]
  {
    float acc = 0.f;
    for (int s = 0; s < S_; ++s) acc += sw[s] * (float)cb[(size_t)s * H_ + tid];
    smix[tid] = acc;
  }
  __syncthreads();

  // out1 = tanh(attn_out @ [mix, q])
  {
    float acc = 0.f;
    const float* ar = attn_out + (size_t)tid * (2 * H_);
    for (int j = 0; j < H_; ++j) acc += ar[j] * smix[j];
    for (int j = 0; j < H_; ++j) acc += ar[H_ + j] * sq[j];
    so1[tid] = tanhf(acc);
  }
  __syncthreads();

  // out2 = relu(lin1 @ out1 + b1)
  {
    float acc = lin1_b[tid];
    const float* lr = lin1_w + (size_t)tid * H_;
    for (int j = 0; j < H_; ++j) acc += lr[j] * so1[j];
    so2[tid] = fmaxf(acc, 0.f);
  }
  __syncthreads();

  // final = lin2 @ out2 + b2  (T=2)
  if (tid < T_) {
    float acc = lin2_b[tid];
    const float* lr = lin2_w + (size_t)tid * H_;
    for (int j = 0; j < H_; ++j) acc += lr[j] * so2[j];
    out[(size_t)b * T_ + tid] = acc;
  }
}

// --------------------------- weight conversion -----------------------------
__global__ void f32_to_bf16_kernel(const float* __restrict__ src,
                                   bf16_t* __restrict__ dst, int n) {
  int i = blockIdx.x * blockDim.x + threadIdx.x;
  if (i < n) dst[i] = (bf16_t)src[i];
}

// ------------------------------- launcher ----------------------------------
extern "C" void kernel_launch(void* const* d_in, const int* in_sizes, int n_in,
                              void* d_out, int out_size, void* d_ws, size_t ws_size,
                              hipStream_t stream) {
  (void)in_sizes; (void)n_in; (void)out_size; (void)ws_size;

  const float* inputs     = (const float*)d_in[0];
  const float* timestamps = (const float*)d_in[1];
  const float* ts_inv     = (const float*)d_in[2];
  const float* Wall_w     = (const float*)d_in[3];
  const float* Wall_b     = (const float*)d_in[4];
  const float* Uall_w     = (const float*)d_in[5];
  const float* Uall_b     = (const float*)d_in[6];
  const float* Wd_w       = (const float*)d_in[7];
  const float* Wd_b       = (const float*)d_in[8];
  const float* gih_w      = (const float*)d_in[9];
  const float* ghh_w      = (const float*)d_in[10];
  const float* gb_ih      = (const float*)d_in[11];
  const float* gb_hh      = (const float*)d_in[12];
  const float* attn_in    = (const float*)d_in[13];
  const float* attn_out   = (const float*)d_in[14];
  const float* Wb         = (const float*)d_in[15];
  const float* wd         = (const float*)d_in[16];
  const float* lin1_w     = (const float*)d_in[17];
  const float* lin1_b     = (const float*)d_in[18];
  const float* lin2_w     = (const float*)d_in[19];
  const float* lin2_b     = (const float*)d_in[20];

  char* ws = (char*)d_ws;
  size_t off = 0;
  auto alloc = [&](size_t bytes) {
    size_t o = off;
    off = (off + bytes + 255) & ~(size_t)255;
    return o;
  };
  bf16_t* Wd_bf   = (bf16_t*)(ws + alloc((size_t)H_ * H_ * 2));
  bf16_t* Wall_bf = (bf16_t*)(ws + alloc((size_t)G4 * H_ * 2));
  bf16_t* Uall_bf = (bf16_t*)(ws + alloc((size_t)G4 * D_ * 2));
  bf16_t* Wih_bf  = (bf16_t*)(ws + alloc((size_t)G3 * H_ * 2));
  bf16_t* Whh_bf  = (bf16_t*)(ws + alloc((size_t)G3 * H_ * 2));
  bf16_t* tl      = (bf16_t*)(ws + alloc((size_t)S_ * B_ * H_ * 2));
  bf16_t* ctx     = (bf16_t*)(ws + alloc((size_t)B_ * S_ * H_ * 2));
  float*  h_n     = (float*)(ws + alloc((size_t)B_ * H_ * 4));

  auto conv = [&](const float* src, bf16_t* dst, int n) {
    f32_to_bf16_kernel<<<(n + 255) / 256, 256, 0, stream>>>(src, dst, n);
  };
  conv(Wd_w,   Wd_bf,   H_ * H_);
  conv(Wall_w, Wall_bf, G4 * H_);
  conv(Uall_w, Uall_bf, G4 * D_);
  conv(gih_w,  Wih_bf,  G3 * H_);
  conv(ghh_w,  Whh_bf,  G3 * H_);

  tlstm_scan_kernel<<<B_ / BT, 512, 0, stream>>>(
      inputs, ts_inv, Wall_bf, Wall_b, Uall_bf, Uall_b, Wd_bf, Wd_b, tl);

  gru_scan_kernel<<<B_ / BT, 512, 0, stream>>>(
      tl, Wih_bf, gb_ih, Whh_bf, gb_hh, ctx, h_n);

  attn_head_kernel<<<B_, 256, 0, stream>>>(
      h_n, ctx, attn_in, attn_out, timestamps, Wb, wd,
      lin1_w, lin1_b, lin2_w, lin2_b, (float*)d_out);
}
